// Decoder_75892072120909
// MI455X (gfx1250) — matrix operand order repaired and checked by reading, not compiled, not run on hardware
//
#include <hip/hip_runtime.h>
#include <hip/hip_bf16.h>
#include <math.h>
#include <stdint.h>

// ---------------- constants ----------------
#define S_TOT   2100      // 15 * (128 + 10 + 2)
#define W_DIM   512
#define BLK     140
#define NFR     15
#define BATCHN  2
#define HEADS   8
#define DH      64
#define LAYERS  8

typedef __attribute__((ext_vector_type(16))) __bf16 v16bf;
typedef __attribute__((ext_vector_type(8)))  float  v8f;
typedef __attribute__((ext_vector_type(4)))  unsigned int v4u;
typedef __attribute__((ext_vector_type(8)))  int v8i;
typedef __attribute__((ext_vector_type(4)))  int v4i;

union AFrag { v16bf v; __bf16 e[16]; };
union CFrag { v8f   v; float  e[8]; };

// A-matrix 16x32 bf16 fragment K index (ISA 7.12.2)
__device__ __forceinline__ int a_k(int i, int lane) {
  int r = i >> 1;
  return ((r >> 2) << 4) + ((lane >> 4) << 3) + ((r & 3) << 1) + (i & 1);
}
// B-matrix 32x16 bf16 fragment K index (same K order as A: lanes 0-15 K=0..7,16..23; lanes 16-31 K=8..15,24..31)
__device__ __forceinline__ int b_k(int i, int lane) {
  return ((i >> 3) << 4) + ((lane >> 4) << 3) + (i & 7);
}

// ---------------- CDNA5 async global->LDS (ASYNCcnt) ----------------
__device__ __forceinline__ void async_load_b128(const void* g, void* l) {
  unsigned lds = (unsigned)(uintptr_t)l;   // addr[31:0] = wave-relative LDS address
  asm volatile("global_load_async_to_lds_b128 %0, %1, off"
               :: "v"(lds), "v"(g) : "memory");
}
__device__ __forceinline__ void wait_async0() {
  asm volatile("s_wait_asynccnt 0x0" ::: "memory");
}

// ---------------- CDNA5 Tensor Data Mover: 2-D bf16 tile -> LDS ----------------
#if __has_builtin(__builtin_amdgcn_tensor_load_to_lds)
#define HAVE_TDM 1
#endif
__device__ __forceinline__ void wait_tensor0() {
#if __has_builtin(__builtin_amdgcn_s_wait_tensorcnt)
  __builtin_amdgcn_s_wait_tensorcnt(0);
#else
  asm volatile("s_wait_tensorcnt 0x0" ::: "memory");
#endif
}
#ifdef HAVE_TDM
// Builds the D# per ISA ch.8: group0 {count=1, lds_addr[63:32], global_addr[120:64], type=2},
// group1 {data_size=2B, tensor_dim0, tensor_dim1 (OOB rows read zero), tile_dim0/1,
//         tensor_dim0_stride}; groups 2/3 zero (2-D tensor).
__device__ __forceinline__ void tdm_load_2d(const void* g, unsigned lds,
                                            unsigned dim0, unsigned dim1,
                                            unsigned tile0, unsigned tile1,
                                            unsigned stride0) {
  unsigned long long ga = (unsigned long long)(uintptr_t)g;
  v4u g0;
  g0[0] = 1u;                                     // count=1 (valid), user mode
  g0[1] = lds;                                    // lds_addr (bytes)
  g0[2] = (unsigned)ga;                           // global_addr[31:0]
  g0[3] = (unsigned)((ga >> 32) & 0x01FFFFFFull)  // global_addr[56:32]
          | 0x80000000u;                          // type=2 ("image") in bits 127:126
  v8i g1;
  g1[0] = (int)(1u << 16);                        // workgroup_mask=0, data_size=1 (2 bytes)
  g1[1] = (int)((dim0 & 0xFFFFu) << 16);          // tensor_dim0[15:0] -> bits 63:48
  g1[2] = (int)((dim0 >> 16) | ((dim1 & 0xFFFFu) << 16));   // dim0[31:16] | dim1[15:0]
  g1[3] = (int)((dim1 >> 16) | ((tile0 & 0xFFFFu) << 16));  // dim1[31:16] | tile_dim0
  g1[4] = (int)(tile1 & 0xFFFFu);                 // tile_dim1 ; tile_dim2 = 0
  g1[5] = (int)stride0;                           // tensor_dim0_stride[31:0]
  g1[6] = 0;                                      // stride[47:32]=0 ; dim1_stride lo = 0
  g1[7] = 0;
  v4i z4 = {0, 0, 0, 0};
#if defined(__clang_major__) && __clang_major__ >= 23
  v8i z8 = {0, 0, 0, 0, 0, 0, 0, 0};
  __builtin_amdgcn_tensor_load_to_lds(g0, g1, z4, z4, z8, 0);
#else
  __builtin_amdgcn_tensor_load_to_lds(g0, g1, z4, z4, 0);
#endif
}
#endif

// ---------------- fp32 -> bf16 convert ----------------
__global__ void cvt_bf16_kernel(const float* __restrict__ src, __bf16* __restrict__ dst, int n) {
  int i = blockIdx.x * blockDim.x + threadIdx.x;
  if (i < n) dst[i] = (__bf16)src[i];
}

// ---------------- token assembly + sinusoidal pos emb ----------------
__global__ void embed_kernel(const float* __restrict__ x, const float* __restrict__ f,
                             const float* __restrict__ delim, float* __restrict__ h) {
  int idx = blockIdx.x * blockDim.x + threadIdx.x;
  const int total = BATCHN * S_TOT * W_DIM;
  if (idx >= total) return;
  int c = idx & (W_DIM - 1);
  int s = (idx >> 9) % S_TOT;
  int b = idx / (W_DIM * S_TOT);
  int i = s / BLK, r = s - i * BLK;
  float val;
  if (r < 128)       val = x[(((size_t)b * NFR + i) * 128 + r) * W_DIM + c];
  else if (r == 128) val = delim[c];
  else if (r < 139)  val = f[(((size_t)b * NFR + i) * 10 + (r - 129)) * W_DIM + c];
  else               val = delim[c];
  int k2 = c >> 1;
  float den = __expf((float)(2 * k2) * (9.210340371976184f / 512.0f)); // 10000^(2k/512)
  float ang = (float)s / den;
  val += (c & 1) ? __cosf(ang) : __sinf(ang);
  h[idx] = val;
}

// ---------------- layernorm (one row per block, 256 threads) -> bf16 ----------------
__global__ void ln_bf16_kernel(const float* __restrict__ h, const float* __restrict__ g,
                               const float* __restrict__ bb, __bf16* __restrict__ out) {
  int row = blockIdx.x;
  int tid = threadIdx.x;
  const float* hp = h + (size_t)row * W_DIM;
  float x0 = hp[tid], x1 = hp[tid + 256];
  __shared__ float red[8];
  float s = x0 + x1;
  for (int m = 16; m; m >>= 1) s += __shfl_xor(s, m, 32);
  if ((tid & 31) == 0) red[tid >> 5] = s;
  __syncthreads();
  float mu = 0.f;
  for (int i = 0; i < 8; ++i) mu += red[i];
  mu *= (1.0f / W_DIM);
  __syncthreads();
  float d0 = x0 - mu, d1 = x1 - mu;
  float vs = d0 * d0 + d1 * d1;
  for (int m = 16; m; m >>= 1) vs += __shfl_xor(vs, m, 32);
  if ((tid & 31) == 0) red[tid >> 5] = vs;
  __syncthreads();
  float var = 0.f;
  for (int i = 0; i < 8; ++i) var += red[i];
  var *= (1.0f / W_DIM);
  float rstd = rsqrtf(var + 1e-5f);
  out[(size_t)row * W_DIM + tid]       = (__bf16)(d0 * rstd * g[tid] + bb[tid]);
  out[(size_t)row * W_DIM + tid + 256] = (__bf16)(d1 * rstd * g[tid + 256] + bb[tid + 256]);
}

// ---------------- WMMA GEMM: C[M,N] = A[M,K](bf16) @ W[N,K]^T(bf16) + bias ----------------
// modes: 0 = fp32 store, 1 = bf16 store, 2 = gelu->bf16, 3 = fp32 residual add (out = res + C)
// 256 threads = 8 waves; block tile 128x128; wave tile 16x128 (8 WMMAs / K-step).
// Double-buffered LDS; tiles staged by the Tensor Data Mover (wave 0 issues two
// tensor_load_to_lds per K-step, TENSORcnt-waited), overlapped with WMMA.
#define BM 128
#define BN 128
#define BK 32
__global__ void gemm_bf16_kernel(const __bf16* __restrict__ A, const __bf16* __restrict__ Wt,
                                 const float* __restrict__ bias, float* __restrict__ outf,
                                 __bf16* __restrict__ outh, const float* __restrict__ res,
                                 int M, int N, int K, int mode) {
  __shared__ __align__(16) __bf16 sA[2][BM * BK];
  __shared__ __align__(16) __bf16 sB[2][BN * BK];
  int tid = threadIdx.x;
  int w = tid >> 5, lane = tid & 31;
  int m0 = blockIdx.x * BM, n0 = blockIdx.y * BN;
  bool issuer = (tid < 32);   // wave 0 drives the TDM

  CFrag acc[8];
  for (int nt = 0; nt < 8; ++nt)
    for (int v = 0; v < 8; ++v) acc[nt].e[v] = 0.f;

#ifdef HAVE_TDM
  if (issuer) {
    tdm_load_2d(&A[(size_t)m0 * K], (unsigned)(uintptr_t)&sA[0][0],
                (unsigned)K, (unsigned)(M - m0), BK, BM, (unsigned)K);
    tdm_load_2d(&Wt[(size_t)n0 * K], (unsigned)(uintptr_t)&sB[0][0],
                (unsigned)K, (unsigned)(N - n0), BK, BN, (unsigned)K);
    wait_tensor0();
  }
  __syncthreads();
#else
  int r0 = tid >> 2, kg0 = (tid & 3) << 3;
  int r1 = r0 + 64;
  int ga0 = min(m0 + r0, M - 1), ga1 = min(m0 + r1, M - 1);
  async_load_b128(&A[(size_t)ga0 * K + kg0], &sA[0][r0 * BK + kg0]);
  async_load_b128(&A[(size_t)ga1 * K + kg0], &sA[0][r1 * BK + kg0]);
  async_load_b128(&Wt[(size_t)(n0 + r0) * K + kg0], &sB[0][r0 * BK + kg0]);
  async_load_b128(&Wt[(size_t)(n0 + r1) * K + kg0], &sB[0][r1 * BK + kg0]);
  wait_async0();
  __syncthreads();
#endif

  int buf = 0;
  for (int k0 = 0; k0 < K; k0 += BK) {
    int kn = k0 + BK;
    if (kn < K) {  // stage next tile into the other buffer (DMA overlaps WMMA below)
#ifdef HAVE_TDM
      if (issuer) {
        tdm_load_2d(&A[(size_t)m0 * K + kn], (unsigned)(uintptr_t)&sA[buf ^ 1][0],
                    (unsigned)K, (unsigned)(M - m0), BK, BM, (unsigned)K);
        tdm_load_2d(&Wt[(size_t)n0 * K + kn], (unsigned)(uintptr_t)&sB[buf ^ 1][0],
                    (unsigned)K, (unsigned)(N - n0), BK, BN, (unsigned)K);
      }
#else
      async_load_b128(&A[(size_t)ga0 * K + kn + kg0], &sA[buf ^ 1][r0 * BK + kg0]);
      async_load_b128(&A[(size_t)ga1 * K + kn + kg0], &sA[buf ^ 1][r1 * BK + kg0]);
      async_load_b128(&Wt[(size_t)(n0 + r0) * K + kn + kg0], &sB[buf ^ 1][r0 * BK + kg0]);
      async_load_b128(&Wt[(size_t)(n0 + r1) * K + kn + kg0], &sB[buf ^ 1][r1 * BK + kg0]);
#endif
    }

    AFrag af;
#pragma unroll
    for (int i = 0; i < 16; ++i)
      af.e[i] = sA[buf][((w << 4) + (lane & 15)) * BK + a_k(i, lane)];

#pragma unroll
    for (int g4 = 0; g4 < 2; ++g4) {   // two groups of 4 B-fragments to cap VGPR use
      AFrag bf_[4];
#pragma unroll
      for (int j = 0; j < 4; ++j)
#pragma unroll
        for (int i = 0; i < 16; ++i)
          bf_[j].e[i] = sB[buf][(((g4 * 4 + j) << 4) + (lane & 15)) * BK + b_k(i, lane)];
#pragma unroll
      for (int j = 0; j < 4; ++j)
        acc[g4 * 4 + j].v = __builtin_amdgcn_wmma_f32_16x16x32_bf16(
            false, af.v, false, bf_[j].v, (short)0, acc[g4 * 4 + j].v, false, false);
    }

    if (kn < K) {
#ifdef HAVE_TDM
      if (issuer) wait_tensor0();    // next tile's DMA complete
#else
      wait_async0();
#endif
    }
    __syncthreads();   // all waves done reading current buffer; staged data visible
    buf ^= 1;
  }

  // epilogue (C layout: row = v + 8*(lane/16), col = lane&15)
#pragma unroll
  for (int nt = 0; nt < 8; ++nt) {
#pragma unroll
    for (int v = 0; v < 8; ++v) {
      int m = (w << 4) + v + ((lane >> 4) << 3);
      int gm = m0 + m;
      if (gm >= M) continue;
      int gn = n0 + (nt << 4) + (lane & 15);
      float val = acc[nt].e[v] + (bias ? bias[gn] : 0.f);
      size_t off = (size_t)gm * N + gn;
      if (mode == 0)      outf[off] = val;
      else if (mode == 1) outh[off] = (__bf16)val;
      else if (mode == 2) outh[off] = (__bf16)(0.5f * val * (1.0f + erff(val * 0.70710678118f)));
      else                outf[off] = res[off] + val;
    }
  }
}

// ---------------- block-causal mask (analytic, never materialized) ----------------
__device__ __forceinline__ bool attn_allow(int q, int k) {
  if (q >= S_TOT || k >= S_TOT) return false;
  if (k == q) return true;
  int iq = q / BLK, rq = q - iq * BLK;
  if (rq >= 128) return false;                 // non-frame tokens: diagonal only
  int ik = k / BLK, rk = k - ik * BLK;
  if (ik == iq && rk < 128) return true;       // own frame block
  if (k == iq * BLK - 1) return true;          // preceding delimiter
  if (rk >= 128 && rk < 139 && ik <= iq) return true; // dyn+delim blocks j<=i
  return false;
}

// ---------------- flash attention: 1 wave per (b, h, 16-query tile) ----------------
__global__ void attn_kernel(const __bf16* __restrict__ qkv, __bf16* __restrict__ o) {
  int b = blockIdx.z, hh = blockIdx.y;
  int q0 = blockIdx.x << 4;
  int lane = threadIdx.x;
  __shared__ __align__(16) __bf16 sK[32 * DH];
  __shared__ __align__(16) __bf16 sV[32 * DH];
  __shared__ __align__(16) __bf16 sP[16 * 32];

  // Q fragments (A layout), rows clamped in-range
  AFrag aq[2];
  {
    int m = lane & 15;
    int qs = q0 + m; if (qs >= S_TOT) qs = S_TOT - 1;
    const __bf16* qrow = qkv + ((size_t)b * S_TOT + qs) * 1536 + hh * DH;
#pragma unroll
    for (int c = 0; c < 2; ++c)
#pragma unroll
      for (int i = 0; i < 16; ++i)
        aq[c].e[i] = qrow[c * 32 + a_k(i, lane)];
  }

  // block-causal chunk skipping: furthest key any row of this tile may see
  int kmax = 0;
  for (int r = 0; r < 16; ++r) {
    int q = q0 + r; if (q >= S_TOT) break;
    int iq = q / BLK, rq = q - iq * BLK;
    int rm = (rq < 128) ? (iq * BLK + 138) : q;
    kmax = max(kmax, rm);
  }
  kmax = min(kmax, S_TOT - 1);

  CFrag accO[4];
  for (int nt = 0; nt < 4; ++nt)
    for (int v = 0; v < 8; ++v) accO[nt].e[v] = 0.f;
  float m8[8], l8[8];
  for (int v = 0; v < 8; ++v) { m8[v] = -1e30f; l8[v] = 0.f; }

  int half = lane >> 4, col = lane & 15;

  for (int k0 = 0; k0 <= kmax; k0 += 32) {
    // async-stage K,V (32 keys x 64 dh bf16 each); lane l owns key l's row
    int cs = min(k0 + lane, S_TOT - 1);
    const __bf16* kr = qkv + ((size_t)b * S_TOT + cs) * 1536 + 512 + hh * DH;
    const __bf16* vr = kr + 512;
#pragma unroll
    for (int j = 0; j < 8; ++j) {
      async_load_b128(&kr[j * 8], &sK[lane * DH + j * 8]);
      async_load_b128(&vr[j * 8], &sV[lane * DH + j * 8]);
    }
    wait_async0();
    __syncthreads();

    // scores = Q @ K^T  (2 n-tiles of 16 keys, K-dim = dh in 2 chunks of 32)
    CFrag sc[2];
#pragma unroll
    for (int nt = 0; nt < 2; ++nt) {
      for (int v = 0; v < 8; ++v) sc[nt].e[v] = 0.f;
#pragma unroll
      for (int c = 0; c < 2; ++c) {
        AFrag bk;
#pragma unroll
        for (int i = 0; i < 16; ++i)
          bk.e[i] = sK[(nt * 16 + col) * DH + c * 32 + b_k(i, lane)];
        sc[nt].v = __builtin_amdgcn_wmma_f32_16x16x32_bf16(
            false, aq[c].v, false, bk.v, (short)0, sc[nt].v, false, false);
      }
    }

    // online softmax (row reductions across the 16 lanes sharing each row)
#pragma unroll
    for (int v = 0; v < 8; ++v) {
      int q = q0 + v + half * 8;
      float s0 = sc[0].e[v] * 0.125f; if (!attn_allow(q, k0 + col))      s0 = -1e30f;
      float s1 = sc[1].e[v] * 0.125f; if (!attn_allow(q, k0 + 16 + col)) s1 = -1e30f;
      float rm = fmaxf(s0, s1);
#pragma unroll
      for (int msk = 8; msk; msk >>= 1) rm = fmaxf(rm, __shfl_xor(rm, msk, 16));
      float mnew = fmaxf(m8[v], rm);
      float corr = __expf(m8[v] - mnew);
      float p0 = (s0 > -1e29f) ? __expf(s0 - mnew) : 0.f;
      float p1 = (s1 > -1e29f) ? __expf(s1 - mnew) : 0.f;
      float rs = p0 + p1;
#pragma unroll
      for (int msk = 8; msk; msk >>= 1) rs += __shfl_xor(rs, msk, 16);
      l8[v] = l8[v] * corr + rs;
      m8[v] = mnew;
#pragma unroll
      for (int nt = 0; nt < 4; ++nt) accO[nt].e[v] *= corr;
      int m = v + half * 8;
      sP[m * 32 + col]      = (__bf16)p0;   // transpose C-layout -> row-major via LDS
      sP[m * 32 + 16 + col] = (__bf16)p1;
    }
    __syncthreads();

    // O += P @ V
    AFrag ap;
#pragma unroll
    for (int i = 0; i < 16; ++i)
      ap.e[i] = sP[(lane & 15) * 32 + a_k(i, lane)];
#pragma unroll
    for (int nt = 0; nt < 4; ++nt) {
      AFrag bv;
#pragma unroll
      for (int i = 0; i < 16; ++i)
        bv.e[i] = sV[b_k(i, lane) * DH + nt * 16 + col];
      accO[nt].v = __builtin_amdgcn_wmma_f32_16x16x32_bf16(
          false, ap.v, false, bv.v, (short)0, accO[nt].v, false, false);
    }
    __syncthreads();
  }

  // normalize + store as bf16 for the o-projection GEMM
#pragma unroll
  for (int v = 0; v < 8; ++v) {
    int q = q0 + v + half * 8;
    if (q >= S_TOT) continue;
    float inv = 1.0f / l8[v];
#pragma unroll
    for (int nt = 0; nt < 4; ++nt)
      o[((size_t)b * S_TOT + q) * W_DIM + hh * DH + nt * 16 + col] = (__bf16)(accO[nt].e[v] * inv);
  }
}

// ---------------- gather frame-token rows -> bf16 for logits GEMM ----------------
__global__ void gather_kernel(const float* __restrict__ h, __bf16* __restrict__ g) {
  int idx = blockIdx.x * blockDim.x + threadIdx.x;
  if (idx >= 3840 * W_DIM) return;
  int c = idx & (W_DIM - 1);
  int r = idx >> 9;
  int b = r / 1920, t = r % 1920;
  int i = t >> 7, tok = t & 127;
  int s = i * BLK + tok;
  g[idx] = (__bf16)h[((size_t)b * S_TOT + s) * W_DIM + c];
}

// ---------------- host launch ----------------
extern "C" void kernel_launch(void* const* d_in, const int* in_sizes, int n_in,
                              void* d_out, int out_size, void* d_ws, size_t ws_size,
                              hipStream_t stream) {
  (void)in_sizes; (void)n_in; (void)out_size; (void)ws_size;
  const float* x     = (const float*)d_in[0];
  const float* f     = (const float*)d_in[1];
  const float* delim = (const float*)d_in[2];
  const float* ln1_g = (const float*)d_in[3];
  const float* ln1_b = (const float*)d_in[4];
  const float* wqkv  = (const float*)d_in[5];
  const float* bqkv  = (const float*)d_in[6];
  const float* wo    = (const float*)d_in[7];
  const float* bo    = (const float*)d_in[8];
  const float* ln2_g = (const float*)d_in[9];
  const float* ln2_b = (const float*)d_in[10];
  const float* wfc   = (const float*)d_in[11];
  const float* bfc   = (const float*)d_in[12];
  const float* wproj = (const float*)d_in[13];
  const float* bproj = (const float*)d_in[14];
  const float* predw = (const float*)d_in[15];

  char* ws = (char*)d_ws;
  size_t off = 0;
  auto take = [&](size_t bytes) -> char* {
    char* p = ws + off;
    off += (bytes + 255) & ~(size_t)255;
    return p;
  };

  const size_t nwqkv = (size_t)LAYERS * 1536 * 512;
  const size_t nwo   = (size_t)LAYERS * 512 * 512;
  const size_t nwfc  = (size_t)LAYERS * 2048 * 512;
  const size_t nwpj  = (size_t)LAYERS * 512 * 2048;
  const size_t npred = (size_t)1024 * 512;
  const int rows = BATCHN * S_TOT;  // 4200

  __bf16* wqkv_h = (__bf16*)take(nwqkv * 2);
  __bf16* wo_h   = (__bf16*)take(nwo * 2);
  __bf16* wfc_h  = (__bf16*)take(nwfc * 2);
  __bf16* wpj_h  = (__bf16*)take(nwpj * 2);
  __bf16* pred_h = (__bf16*)take(npred * 2);
  float*  h      = (float*) take((size_t)rows * W_DIM * 4);
  __bf16* abuf   = (__bf16*)take((size_t)rows * W_DIM * 2);
  __bf16* qkvb   = (__bf16*)take((size_t)rows * 1536 * 2);
  __bf16* obuf   = (__bf16*)take((size_t)rows * W_DIM * 2);
  __bf16* mlpb   = (__bf16*)take((size_t)rows * 2048 * 2);
  __bf16* gath   = (__bf16*)take((size_t)3840 * W_DIM * 2);

  auto cvt = [&](const float* src, __bf16* dst, size_t n) {
    cvt_bf16_kernel<<<dim3((unsigned)((n + 255) / 256)), dim3(256), 0, stream>>>(src, dst, (int)n);
  };
  cvt(wqkv,  wqkv_h, nwqkv);
  cvt(wo,    wo_h,   nwo);
  cvt(wfc,   wfc_h,  nwfc);
  cvt(wproj, wpj_h,  nwpj);
  cvt(predw, pred_h, npred);

  embed_kernel<<<dim3((rows * W_DIM + 255) / 256), 256, 0, stream>>>(x, f, delim, h);

  auto gemm = [&](const __bf16* A, const __bf16* Wt, const float* bias, float* of,
                  __bf16* oh, const float* res, int M, int N, int K, int mode) {
    gemm_bf16_kernel<<<dim3((M + BM - 1) / BM, N / BN), dim3(256), 0, stream>>>(
        A, Wt, bias, of, oh, res, M, N, K, mode);
  };

  for (int l = 0; l < LAYERS; ++l) {
    ln_bf16_kernel<<<rows, 256, 0, stream>>>(h, ln1_g + l * 512, ln1_b + l * 512, abuf);
    gemm(abuf, wqkv_h + (size_t)l * 1536 * 512, bqkv + (size_t)l * 1536,
         nullptr, qkvb, nullptr, rows, 1536, 512, 1);
    attn_kernel<<<dim3((S_TOT + 15) / 16, HEADS, BATCHN), 32, 0, stream>>>(qkvb, obuf);
    gemm(obuf, wo_h + (size_t)l * 512 * 512, bo + l * 512,
         h, nullptr, h, rows, 512, 512, 3);
    ln_bf16_kernel<<<rows, 256, 0, stream>>>(h, ln2_g + l * 512, ln2_b + l * 512, abuf);
    gemm(abuf, wfc_h + (size_t)l * 2048 * 512, bfc + (size_t)l * 2048,
         nullptr, mlpb, nullptr, rows, 2048, 512, 2);
    gemm(mlpb, wpj_h + (size_t)l * 512 * 2048, bproj + l * 512,
         h, nullptr, h, rows, 512, 2048, 3);
  }

  gather_kernel<<<dim3((3840 * W_DIM + 255) / 256), 256, 0, stream>>>(h, gath);
  gemm(gath, pred_h, nullptr, (float*)d_out, nullptr, nullptr, 3840, 1024, 512, 0);
}
